// DARNNModel_80109730005450
// MI455X (gfx1250) — matrix operand order repaired and checked by reading, not compiled, not run on hardware
//
#include <hip/hip_runtime.h>
#include <hip/hip_bf16.h>

// ---------------------------------------------------------------------------
// DA-RNN fused persistent kernel for MI455X (gfx1250, wave32, WMMA).
// One workgroup (256 threads = 8 waves) owns a 16-row batch tile end-to-end:
//   phase 1: encoder scan over T=128 (input attention + GRU), Henc -> d_ws
//   phase 2: fc_init + decoder GRU
//   phase 3: temporal attention over Henc + output projection
// GEMMs: v_wmma_f32_16x16x32_bf16. tanh: hardware v_tanh_f32 (TRANS32).
// LDS staging of x_t / Henc tiles: global_load_async_to_lds_b128 (ASYNCcnt).
// ---------------------------------------------------------------------------

typedef __bf16 bf16;
typedef __attribute__((ext_vector_type(16))) __bf16 v16bf;
typedef __attribute__((ext_vector_type(8)))  float  v8f;
typedef __attribute__((ext_vector_type(4)))  int    i32x4;

namespace {
constexpr int nB = 2048, nT = 128, nF = 64, nH = 128, nD = 128, nA = 64;
constexpr int ROWS = 16;          // batch rows per workgroup == WMMA M
constexpr int NTHREADS = 256;     // 8 waves (wave32)
constexpr int NWAVES = NTHREADS / 32;
}

// ---- async LDS staging (CDNA5 ASYNCcnt path), guarded by __has_builtin ----
#if defined(__has_builtin)
#  if __has_builtin(__builtin_amdgcn_global_load_async_to_lds_b128) && \
      __has_builtin(__builtin_amdgcn_s_wait_asynccnt)
#    define USE_ASYNC_LDS 1
#  endif
#endif
#ifndef USE_ASYNC_LDS
#  define USE_ASYNC_LDS 0
#endif

#if USE_ASYNC_LDS
// builtin signature (from clang diagnostic): arg0 = int4* in addrspace(1),
// arg1 = LDS destination, arg2/arg3 = imm offset / cache-policy.
__device__ __forceinline__ void async_cp16(void* lds_dst, const void* gsrc) {
  __builtin_amdgcn_global_load_async_to_lds_b128(
      (__attribute__((address_space(1))) i32x4*)const_cast<void*>(gsrc),
      (__attribute__((address_space(3))) i32x4*)lds_dst,
      0, 0);
}
__device__ __forceinline__ void async_wait0() {
  __builtin_amdgcn_s_wait_asynccnt(0);
}
#endif

// ---- hardware tanh (gfx1250 TRANS32 op); delay hint per ISA §7.4/§8 ----
__device__ __forceinline__ float fast_tanh(float v) {
  float r;
  asm("v_tanh_f32 %0, %1\n\t"
      "s_delay_alu instid0(TRANS32_DEP_1)"
      : "=v"(r) : "v"(v));
  return r;
}

__device__ __forceinline__ float sigm(float v) {
  return 1.0f / (1.0f + __expf(-v));   // v_exp_f32 based
}

// ---- WMMA fragment builders (contiguous k = 16h + s, same order in A and B) ----

// A-matrix 16x32 bf16, source LDS row-major [16][lda].
__device__ __forceinline__ v16bf frag_a(const bf16* A, int lda, int k0) {
  const int lane = threadIdx.x & 31;
  const int m  = lane & 15;
  const int kb = k0 + ((lane >> 4) << 4);          // +16 for upper half-wave
  const bf16* p = A + m * lda + kb;
  v16bf r;
  *((float4*)&r)       = *(const float4*)(p);       // K = kb   .. kb+7
  *(((float4*)&r) + 1) = *(const float4*)(p + 8);   // K = kb+8 .. kb+15
  return r;
}

// B-matrix 32x16 bf16 from weight stored row-major [N][ldk] (W.T GEMM).
__device__ __forceinline__ v16bf frag_b(const bf16* W, int ldk, int n0, int k0) {
  const int lane = threadIdx.x & 31;
  const int n  = n0 + (lane & 15);
  const int kb = k0 + ((lane >> 4) << 4);          // +16 for upper half-wave
  const bf16* p = W + n * ldk + kb;
  v16bf r;
  *((float4*)&r)       = *(const float4*)(p);       // K = kb   .. kb+7
  *(((float4*)&r) + 1) = *(const float4*)(p + 8);   // K = kb+8 .. kb+15
  return r;
}

__device__ __forceinline__ v8f wmma_bf16(v16bf a, v16bf b, v8f c) {
  // (neg_a, A, neg_b, B, c_mod, C, reuse_a, reuse_b)
  return __builtin_amdgcn_wmma_f32_16x16x32_bf16(false, a, false, b,
                                                 (short)0, c, false, false);
}

// D[16, n0:n0+16] = A[16,K] @ W[n0:n0+16, K]^T  (K-accumulate, f32 acc)
__device__ __forceinline__ v8f gemm16(const bf16* A, int lda,
                                      const bf16* W, int ldk,
                                      int n0, int K) {
  v8f c = {0.f, 0.f, 0.f, 0.f, 0.f, 0.f, 0.f, 0.f};
  for (int k0 = 0; k0 < K; k0 += 32)
    c = wmma_bf16(frag_a(A, lda, k0), frag_b(W, ldk, n0, k0), c);
  return c;
}

// Store 16x16 f32 C/D tile to LDS, optional per-column bias.
__device__ __forceinline__ void store16(float* C, int ldc, int n0, v8f c,
                                        const float* bias) {
  const int lane = threadIdx.x & 31;
  const int n  = n0 + (lane & 15);
  const int mb = (lane >> 4) << 3;
  const float bv = bias ? bias[n] : 0.0f;
#pragma unroll
  for (int r = 0; r < 8; ++r) C[(mb + r) * ldc + n] = c[r] + bv;
}

// ---- cooperative weight staging: f32 global -> LDS (bf16 or f32) ----
__device__ __forceinline__ void cpw_bf(bf16* dst, const float* src, int n) {
  for (int i = threadIdx.x; i < n; i += NTHREADS) dst[i] = (bf16)src[i];
}
__device__ __forceinline__ void cpw_f(float* dst, const float* src, int n) {
  for (int i = threadIdx.x; i < n; i += NTHREADS) dst[i] = src[i];
}

// ---------------------------------------------------------------------------

__global__ __launch_bounds__(NTHREADS)
void darnn_fused(const float* __restrict__ x,
                 const float* We_ih, const float* We_hh,
                 const float* be_ih, const float* be_hh,
                 const float* W_e,  const float* U_e, const float* b_e,
                 const float* v_e,
                 const float* Wi,   const float* bi,
                 const float* Wd_ih,const float* Wd_hh,
                 const float* bd_ih,const float* bd_hh,
                 const float* W_d,  const float* U_d, const float* v_d,
                 const float* Wo,   const float* bo,  const float* y0,
                 bf16* __restrict__ henc,   // [nB/ROWS][nT][ROWS][nH] bf16 ws
                 float* __restrict__ out) { // [nB]
  const int tid = threadIdx.x;
  const int wid = tid >> 5;
  const int b0  = blockIdx.x * ROWS;

  // -- phase-swapped weight region (max of encoder/decoder footprints) --
  __shared__ __align__(16) unsigned char wraw[183552];
  // -- persistent state --
  __shared__ __align__(16) float h_f[ROWS * nH];      // h (f32 master)
  __shared__ __align__(16) bf16  h_b[ROWS * nH];      // h (bf16 GEMM copy)
  __shared__ __align__(16) float xt [ROWS * nF];      // x_t tile
  __shared__ __align__(16) bf16  xa [ROWS * nF];      // alpha * x_t (bf16)
  __shared__ __align__(16) float hp [ROWS * nA];      // h@W_e^T / d@W_d^T
  __shared__ __align__(16) float gi [ROWS * 3 * nH];  // GRU input gates
  __shared__ __align__(16) float gh [ROWS * 3 * nH];  // GRU hidden gates
  __shared__ __align__(16) float sc [ROWS * nT];      // scores (F or T wide)
  __shared__ __align__(16) float d_f[ROWS * nD];
  __shared__ __align__(16) bf16  d_b[ROWS * nD];
  __shared__ __align__(16) float pbuf[ROWS * 16];     // partial reductions

  // ---------------- phase 1: encoder weights into LDS ----------------
  bf16*  sWeih = (bf16*) (wraw);            // [384][64]
  bf16*  sWehh = (bf16*) (wraw + 49152);    // [384][128]
  bf16*  sWe   = (bf16*) (wraw + 147456);   // [64][128]
  bf16*  sUe   = (bf16*) (wraw + 163840);   // [64][64]
  bf16*  sbe   = (bf16*) (wraw + 172032);   // [64][64]
  float* sve   = (float*)(wraw + 180224);   // [64]
  float* sbeih = (float*)(wraw + 180480);   // [384]
  float* sbehh = (float*)(wraw + 182016);   // [384]

  cpw_bf(sWeih, We_ih, 3 * nH * nF);
  cpw_bf(sWehh, We_hh, 3 * nH * nH);
  cpw_bf(sWe,   W_e,   nA * nH);
  cpw_bf(sUe,   U_e,   nF * nA);
  cpw_bf(sbe,   b_e,   nF * nA);
  cpw_f (sve,   v_e,   nA);
  cpw_f (sbeih, be_ih, 3 * nH);
  cpw_f (sbehh, be_hh, 3 * nH);
  for (int i = tid; i < ROWS * nH; i += NTHREADS) { h_f[i] = 0.0f; h_b[i] = (bf16)0.0f; }
  __syncthreads();

  // ---------------- encoder scan over T ----------------
  for (int t = 0; t < nT; ++t) {
    // stage x_t tile [16][64] into LDS (async overlaps with h-proj GEMM)
#if USE_ASYNC_LDS
    {
      const int b = tid >> 4, w = tid & 15;   // 16 x 16B chunks per row
      const float* gp = x + ((size_t)(b0 + b) * nT + t) * nF + w * 4;
      async_cp16((char*)xt + tid * 16, gp);
    }
#else
    for (int i = tid; i < ROWS * nF; i += NTHREADS) {
      const int b = i >> 6, f = i & (nF - 1);
      xt[i] = x[((size_t)(b0 + b) * nT + t) * nF + f];
    }
#endif
    if (t + 1 < nT && tid < ROWS)   // global_prefetch_b8 of next step's rows
      __builtin_prefetch(x + ((size_t)(b0 + tid) * nT + (t + 1)) * nF, 0, 0);

    // h_proj[16,64] = h @ W_e^T   (4 N-tiles, K=128) -- independent of x_t
    for (int nt = wid; nt < nA / 16; nt += NWAVES)
      store16(hp, nA, nt * 16, gemm16(h_b, nH, sWe, nH, nt * 16, nH), nullptr);
#if USE_ASYNC_LDS
    async_wait0();
#endif
    __syncthreads();

    // input-attention scores: s[b,f] = sum_a v_e[a]*tanh(hp+x*U_e+b_e)
    for (int i = tid; i < ROWS * nF; i += NTHREADS) {
      const int b = i >> 6, f = i & (nF - 1);
      const float xv = xt[b * nF + f];
      float s = 0.0f;
      for (int a = 0; a < nA; ++a)
        s += sve[a] * fast_tanh(hp[b * nA + a] + xv * (float)sUe[f * nA + a]
                                + (float)sbe[f * nA + a]);
      sc[b * nT + f] = s;
    }
    __syncthreads();

    // softmax over F per row, then xa = alpha * x_t (bf16)
    if (tid < ROWS) {
      const int b = tid;
      float mx = -1e30f;
      for (int f = 0; f < nF; ++f) mx = fmaxf(mx, sc[b * nT + f]);
      float sum = 0.0f;
      for (int f = 0; f < nF; ++f) {
        const float e = __expf(sc[b * nT + f] - mx);
        sc[b * nT + f] = e; sum += e;
      }
      const float inv = 1.0f / sum;
      for (int f = 0; f < nF; ++f)
        xa[b * nF + f] = (bf16)(sc[b * nT + f] * inv * xt[b * nF + f]);
    }
    __syncthreads();

    // gi[16,384] = xa @ We_ih^T + be_ih  (24 tiles, K=64)
    for (int nt = wid; nt < (3 * nH) / 16; nt += NWAVES)
      store16(gi, 3 * nH, nt * 16, gemm16(xa, nF, sWeih, nF, nt * 16, nF), sbeih);
    // gh[16,384] = h @ We_hh^T + be_hh   (24 tiles, K=128)
    for (int nt = wid; nt < (3 * nH) / 16; nt += NWAVES)
      store16(gh, 3 * nH, nt * 16, gemm16(h_b, nH, sWehh, nH, nt * 16, nH), sbehh);
    __syncthreads();

    // GRU combine, update h, spill Henc (bf16) to workspace
    for (int i = tid; i < ROWS * nH; i += NTHREADS) {
      const int b = i >> 7, j = i & (nH - 1);
      const float r  = sigm(gi[b * 384 + j]        + gh[b * 384 + j]);
      const float z  = sigm(gi[b * 384 + nH + j]   + gh[b * 384 + nH + j]);
      const float nn = fast_tanh(gi[b * 384 + 2*nH + j] + r * gh[b * 384 + 2*nH + j]);
      const float hv = (1.0f - z) * nn + z * h_f[i];
      h_f[i] = hv;
      h_b[i] = (bf16)hv;
      henc[(((size_t)blockIdx.x * nT + t) * ROWS) * nH + i] = (bf16)hv;
    }
    __syncthreads();
  }

  // ---------------- phase 2: decoder weights into same LDS region -----
  bf16*  sWi   = (bf16*) (wraw);            // [128][128]
  bf16*  sWdhh = (bf16*) (wraw + 32768);    // [384][128]
  bf16*  sWd   = (bf16*) (wraw + 131072);   // [64][128]
  bf16*  sUd   = (bf16*) (wraw + 147456);   // [64][128]
  float* svd   = (float*)(wraw + 163840);   // [64]
  float* sbdih = (float*)(wraw + 164096);   // [384]
  float* sbdhh = (float*)(wraw + 165632);   // [384]
  float* sWdih = (float*)(wraw + 167168);   // [384]  (Wd_ih column)
  float* sWo   = (float*)(wraw + 168704);   // [256]
  float* sbi   = (float*)(wraw + 169728);   // [128]

  cpw_bf(sWi,   Wi,    nD * nH);
  cpw_bf(sWdhh, Wd_hh, 3 * nD * nD);
  cpw_bf(sWd,   W_d,   nA * nD);
  cpw_bf(sUd,   U_d,   nA * nH);
  cpw_f (svd,   v_d,   nA);
  cpw_f (sbdih, bd_ih, 3 * nD);
  cpw_f (sbdhh, bd_hh, 3 * nD);
  cpw_f (sWdih, Wd_ih, 3 * nD);
  cpw_f (sWo,   Wo,    nD + nH);
  cpw_f (sbi,   bi,    nD);
  __syncthreads();

  // d0 = h_T @ Wi^T + bi   (8 tiles, K=128)
  for (int nt = wid; nt < nD / 16; nt += NWAVES)
    store16(d_f, nD, nt * 16, gemm16(h_b, nH, sWi, nH, nt * 16, nH), sbi);
  __syncthreads();
  for (int i = tid; i < ROWS * nD; i += NTHREADS) d_b[i] = (bf16)d_f[i];
  __syncthreads();

  // decoder GRU: gh = d0 @ Wd_hh^T + bd_hh ; gi = y0*Wd_ih + bd_ih
  for (int nt = wid; nt < (3 * nD) / 16; nt += NWAVES)
    store16(gh, 3 * nD, nt * 16, gemm16(d_b, nD, sWdhh, nD, nt * 16, nD), sbdhh);
  __syncthreads();
  const float y0v = y0[0];
  for (int i = tid; i < ROWS * nD; i += NTHREADS) {
    const int b = i >> 7, j = i & (nD - 1);
    const float ir  = y0v * sWdih[j]          + sbdih[j];
    const float iz  = y0v * sWdih[nD + j]     + sbdih[nD + j];
    const float inn = y0v * sWdih[2 * nD + j] + sbdih[2 * nD + j];
    const float r  = sigm(ir + gh[b * 384 + j]);
    const float z  = sigm(iz + gh[b * 384 + nD + j]);
    const float nn = fast_tanh(inn + r * gh[b * 384 + 2 * nD + j]);
    const float dv = (1.0f - z) * nn + z * d_f[i];
    d_f[i] = dv; d_b[i] = (bf16)dv;
  }
  __syncthreads();

  // d_proj[16,64] = d @ W_d^T
  for (int nt = wid; nt < nA / 16; nt += NWAVES)
    store16(hp, nA, nt * 16, gemm16(d_b, nD, sWd, nD, nt * 16, nD), nullptr);
  __syncthreads();

  // ---------------- phase 3: temporal attention ----------------
  float* hp2 = gh;   // reuse as [16][64] H_proj buffer
  for (int t = 0; t < nT; ++t) {
    const bf16* hsrc = henc + (((size_t)blockIdx.x * nT + t) * ROWS) * nH;
#if USE_ASYNC_LDS
    async_cp16((char*)h_b + tid * 16, (const char*)hsrc + tid * 16);
    async_wait0();
#else
    for (int i = tid; i < ROWS * nH; i += NTHREADS) h_b[i] = hsrc[i];
#endif
    __syncthreads();
    // H_proj_t[16,64] = Henc_t @ U_d^T  (4 tiles, K=128)
    for (int nt = wid; nt < nA / 16; nt += NWAVES)
      store16(hp2, nA, nt * 16, gemm16(h_b, nH, sUd, nH, nt * 16, nH), nullptr);
    __syncthreads();
    { // score[b,t] = sum_a v_d[a]*tanh(d_proj + H_proj); 16-way partial/row
      const int b = tid & 15, part = tid >> 4;   // part in 0..15
      float s = 0.0f;
      for (int a = part * 4; a < part * 4 + 4; ++a)
        s += svd[a] * fast_tanh(hp[b * nA + a] + hp2[b * nA + a]);
      pbuf[b * 16 + part] = s;
    }
    __syncthreads();
    if (tid < ROWS) {
      float s = 0.0f;
      for (int p = 0; p < 16; ++p) s += pbuf[tid * 16 + p];
      sc[tid * nT + t] = s;
    }
    __syncthreads();
  }

  // softmax over T per row -> beta in sc
  if (tid < ROWS) {
    float mx = -1e30f;
    for (int t = 0; t < nT; ++t) mx = fmaxf(mx, sc[tid * nT + t]);
    float sum = 0.0f;
    for (int t = 0; t < nT; ++t) {
      const float e = __expf(sc[tid * nT + t] - mx);
      sc[tid * nT + t] = e; sum += e;
    }
    const float inv = 1.0f / sum;
    for (int t = 0; t < nT; ++t) sc[tid * nT + t] *= inv;
  }
  __syncthreads();

  // context[b,:] = sum_t beta[b,t]*Henc[b,t,:]  (per-row weights -> VALU FMA)
  // thread owns column hh for 8 rows (half = upper/lower row group)
  const int half = tid >> 7;        // 0 or 1
  const int hh   = tid & 127;
  float acc[8] = {0.f, 0.f, 0.f, 0.f, 0.f, 0.f, 0.f, 0.f};
  for (int t = 0; t < nT; ++t) {
    const bf16* hsrc = henc + (((size_t)blockIdx.x * nT + t) * ROWS) * nH;
#if USE_ASYNC_LDS
    async_cp16((char*)h_b + tid * 16, (const char*)hsrc + tid * 16);
    async_wait0();
#else
    for (int i = tid; i < ROWS * nH; i += NTHREADS) h_b[i] = hsrc[i];
#endif
    __syncthreads();
#pragma unroll
    for (int j = 0; j < 8; ++j) {
      const int row = half * 8 + j;
      acc[j] += sc[row * nT + t] * (float)h_b[row * nH + hh];
    }
    __syncthreads();
  }
  float* ctxf = gi;  // reuse as [16][128]
#pragma unroll
  for (int j = 0; j < 8; ++j) ctxf[(half * 8 + j) * nH + hh] = acc[j];
  __syncthreads();

  // out[b] = Wo[0:128].d[b] + Wo[128:256].ctx[b] + bo
  {
    const int b = tid & 15, part = tid >> 4;   // 16 parts x 16 elems = 256
    float s = 0.0f;
    for (int jj = part * 16; jj < part * 16 + 16; ++jj) {
      const float w = sWo[jj];
      s += (jj < nD) ? w * d_f[b * nD + jj]
                     : w * ctxf[b * nH + (jj - nD)];
    }
    pbuf[b * 16 + part] = s;
  }
  __syncthreads();
  if (tid < ROWS) {
    float s = 0.0f;
    for (int p = 0; p < 16; ++p) s += pbuf[tid * 16 + p];
    out[b0 + tid] = s + bo[0];
  }
}

// ---------------------------------------------------------------------------

extern "C" void kernel_launch(void* const* d_in, const int* in_sizes, int n_in,
                              void* d_out, int out_size, void* d_ws, size_t ws_size,
                              hipStream_t stream) {
  const float* x     = (const float*)d_in[0];
  const float* We_ih = (const float*)d_in[1];
  const float* We_hh = (const float*)d_in[2];
  const float* be_ih = (const float*)d_in[3];
  const float* be_hh = (const float*)d_in[4];
  const float* W_e   = (const float*)d_in[5];
  const float* U_e   = (const float*)d_in[6];
  const float* b_e   = (const float*)d_in[7];
  const float* v_e   = (const float*)d_in[8];
  const float* Wi    = (const float*)d_in[9];
  const float* bi    = (const float*)d_in[10];
  const float* Wd_ih = (const float*)d_in[11];
  const float* Wd_hh = (const float*)d_in[12];
  const float* bd_ih = (const float*)d_in[13];
  const float* bd_hh = (const float*)d_in[14];
  const float* W_d   = (const float*)d_in[15];
  const float* U_d   = (const float*)d_in[16];
  const float* v_d   = (const float*)d_in[17];
  const float* Wo    = (const float*)d_in[18];
  const float* bo    = (const float*)d_in[19];
  const float* y0    = (const float*)d_in[20];

  // Henc workspace: B*T*H bf16 = 64 MiB
  bf16* henc = (bf16*)d_ws;

  dim3 grid(nB / ROWS);      // 128 workgroups, each owns 16 batch rows
  dim3 block(NTHREADS);      // 8 waves (wave32)
  hipLaunchKernelGGL(darnn_fused, grid, block, 0, stream,
                     x, We_ih, We_hh, be_ih, be_hh, W_e, U_e, b_e, v_e,
                     Wi, bi, Wd_ih, Wd_hh, bd_ih, bd_hh, W_d, U_d, v_d,
                     Wo, bo, y0, henc, (float*)d_out);
}
